// LowRankMultiheadAttention_37967510896874
// MI455X (gfx1250) — matrix operand order repaired and checked by reading, not compiled, not run on hardware
//
#include <hip/hip_runtime.h>

#define DEV __device__ __forceinline__

typedef _Float16 v16h __attribute__((ext_vector_type(16)));
typedef _Float16 h4   __attribute__((ext_vector_type(4)));
typedef float    v8f  __attribute__((ext_vector_type(8)));

// ---------------- problem constants ----------------
constexpr int B_  = 4;
constexpr int TQ  = 1024;
constexpr int TKV = 1024;
constexpr int NPR = 5;
constexpr int H_  = 16;
constexpr int D_  = 64;
constexpr int IN_ = 1024;
constexpr int OUT_= 1024;        // H*D
constexpr int R_  = 64;
constexpr int KV_ = NPR + TKV;   // 1029
constexpr int KVP = 1056;        // KV padded to multiple of 32 (33*32)

DEV int laneid() { return threadIdx.x & 31; }

// ---------------- CDNA5 async global->LDS staging ----------------
// GLOBAL_LOAD_ASYNC_TO_LDS_B128: vdst = VGPR holding LDS byte address,
// vaddr = 64-bit global address. Tracked with ASYNCcnt.
DEV void async_load_b128(unsigned lds_off, const void* gptr) {
  asm volatile("global_load_async_to_lds_b128 %0, %1, off"
               :: "v"(lds_off), "v"(gptr) : "memory");
}
DEV void wait_asynccnt0() {
  asm volatile("s_wait_asynccnt 0x0" ::: "memory");
}
DEV unsigned lds_off32(const void* p) { return (unsigned)(uintptr_t)p; }

// ---------------- WMMA fragment loaders ----------------
// A fragment (16x32 f16) from row-major memory, row stride `ld` halves.
// lane<16: row m0+lane, K={0..15}; lane>=16: row m0+lane-16,
// K={16..31} -> two contiguous 16B loads per lane.
DEV v16h load_afrag(const _Float16* base, int m0, int k0, int ld) {
  int lane = laneid();
  int row  = m0 + (lane & 15);
  int koff = (lane >> 4) * 16;
  const _Float16* p = base + (size_t)row * ld + k0 + koff;
  union { v16h v; uint4 u[2]; } r;
  r.u[0] = *reinterpret_cast<const uint4*>(p);
  r.u[1] = *reinterpret_cast<const uint4*>(p + 8);
  return r.v;
}

// B fragment (32x16 f16) where B[k][n] = Mem[n][k] (row-major in n, K
// contiguous) -> one 32B contiguous load per lane. Used for K^T and V^T.
DEV v16h load_bfrag_t(const _Float16* base, int n0, int k0, int ld) {
  int lane = laneid();
  int n    = n0 + (lane & 15);
  int koff = (lane >> 4) * 16;
  const _Float16* p = base + (size_t)n * ld + k0 + koff;
  union { v16h v; uint4 u[2]; } r;
  r.u[0] = *reinterpret_cast<const uint4*>(p);
  r.u[1] = *reinterpret_cast<const uint4*>(p + 8);
  return r.v;
}

// B fragment from pre-swizzled (fragment-linear) weight tile.
DEV v16h load_bfrag_swz(const _Float16* tile) {
  const _Float16* p = tile + laneid() * 16;
  union { v16h v; uint4 u[2]; } r;
  r.u[0] = *reinterpret_cast<const uint4*>(p);
  r.u[1] = *reinterpret_cast<const uint4*>(p + 8);
  return r.v;
}

// C/D layout helpers.
DEV int c_row(int r) { return r + ((laneid() >> 4) << 3); }
DEV int c_col()      { return laneid() & 15; }

DEV v8f wmma_f16(v16h a, v16h b, v8f c) {
  return __builtin_amdgcn_wmma_f32_16x16x32_f16(false, a, false, b, (short)0, c,
                                                false, false);
}

// ---------------- weight swizzle: f32 [K][N] -> f16 fragment-linear ----------------
__global__ void k_swizzle_w(const float* __restrict__ W, _Float16* __restrict__ dst,
                            int K, int N) {
  int e = blockIdx.x * blockDim.x + threadIdx.x;
  if (e >= K * N) return;
  int tile = e >> 9;          // 512 halves per 32x16 tile
  int rr   = e & 511;
  int lane = rr >> 4, i = rr & 15;
  int ntiles = N >> 4;
  int kt = tile / ntiles, nt = tile % ntiles;
  int k, n;
  if (lane < 16) { k = kt * 32 + i;      n = nt * 16 + lane;      }
  else           { k = kt * 32 + 16 + i; n = nt * 16 + lane - 16; }
  dst[e] = (_Float16)W[(size_t)k * N + n];
}

// ---------------- gather + f32->f16 convert ----------------
__global__ void k_gather_xkv(const float* __restrict__ prompt,
                             const float* __restrict__ kvq,
                             const int* __restrict__ task_idx,
                             _Float16* __restrict__ dst) {
  int rg = blockIdx.x;                 // [0, B*KVP)
  int b = rg / KVP, r = rg % KVP;
  int col = threadIdx.x * 4;
  const float* src = nullptr;
  if (r < NPR)      src = prompt + ((size_t)task_idx[b] * NPR + r) * IN_ + col;
  else if (r < KV_) src = kvq    + ((size_t)b * TKV + (r - NPR)) * IN_ + col;
  h4 o = { (_Float16)0, (_Float16)0, (_Float16)0, (_Float16)0 };
  if (src) {
    float4 f = *reinterpret_cast<const float4*>(src);
    o.x = (_Float16)f.x; o.y = (_Float16)f.y; o.z = (_Float16)f.z; o.w = (_Float16)f.w;
  }
  *reinterpret_cast<h4*>(dst + (size_t)rg * IN_ + col) = o;
}

__global__ void k_gather_xpos(const float* __restrict__ pe,
                              const int* __restrict__ indices_b,
                              _Float16* __restrict__ dst) {
  int rg = blockIdx.x;                 // [0, B*KVP)
  int b = rg / KVP, r = rg % KVP;
  int col = threadIdx.x * 4;
  h4 o = { (_Float16)0, (_Float16)0, (_Float16)0, (_Float16)0 };
  if (r >= NPR && r < KV_) {
    int idx = indices_b[(size_t)b * TKV + (r - NPR)];
    float4 f = *reinterpret_cast<const float4*>(pe + (size_t)idx * OUT_ + col);
    o.x = (_Float16)f.x; o.y = (_Float16)f.y; o.z = (_Float16)f.z; o.w = (_Float16)f.w;
  }
  *reinterpret_cast<h4*>(dst + (size_t)rg * OUT_ + col) = o;
}

__global__ void k_gather_xpea(const float* __restrict__ pe,
                              const int* __restrict__ indices_a,
                              _Float16* __restrict__ dst) {
  int rg = blockIdx.x;                 // [0, B*TQ)
  int col = threadIdx.x * 4;
  int idx = indices_a[rg];
  float4 f = *reinterpret_cast<const float4*>(pe + (size_t)idx * OUT_ + col);
  h4 o = { (_Float16)f.x, (_Float16)f.y, (_Float16)f.z, (_Float16)f.w };
  *reinterpret_cast<h4*>(dst + (size_t)rg * OUT_ + col) = o;
}

// ---------------- GEMM1: T[M,64] = A[M,1024] @ W1(swizzled) ----------------
// One wave = 32 rows (2 m-tiles x 4 n-tiles). k-loop unrolled by 2 with
// ping-pong A registers (no fragment copies), A prefetched one step ahead.
__global__ void __launch_bounds__(128) k_gemm1(const _Float16* __restrict__ A,
                                               const _Float16* __restrict__ Wswz,
                                               _Float16* __restrict__ T, int M) {
  int wid = (blockIdx.x * blockDim.x + threadIdx.x) >> 5;
  int m0 = wid * 32;
  if (m0 >= M) return;
  v8f acc0 = {}, acc1 = {}, acc2 = {}, acc3 = {};
  v8f acc4 = {}, acc5 = {}, acc6 = {}, acc7 = {};
  v16h a0 = load_afrag(A, m0,      0, IN_);
  v16h a1 = load_afrag(A, m0 + 16, 0, IN_);
  for (int kt = 0; kt < IN_ / 32; kt += 2) {
    // even k-step
    const _Float16* wt = Wswz + ((size_t)kt << 11);   // kt * 4 tiles * 512
    v16h b0 = load_bfrag_swz(wt);
    v16h b1 = load_bfrag_swz(wt + 512);
    v16h b2 = load_bfrag_swz(wt + 1024);
    v16h b3 = load_bfrag_swz(wt + 1536);
    v16h a0n = load_afrag(A, m0,      (kt + 1) * 32, IN_);
    v16h a1n = load_afrag(A, m0 + 16, (kt + 1) * 32, IN_);
    acc0 = wmma_f16(a0, b0, acc0);
    acc1 = wmma_f16(a0, b1, acc1);
    acc2 = wmma_f16(a0, b2, acc2);
    acc3 = wmma_f16(a0, b3, acc3);
    acc4 = wmma_f16(a1, b0, acc4);
    acc5 = wmma_f16(a1, b1, acc5);
    acc6 = wmma_f16(a1, b2, acc6);
    acc7 = wmma_f16(a1, b3, acc7);
    // odd k-step
    const _Float16* wt2 = wt + 2048;
    b0 = load_bfrag_swz(wt2);
    b1 = load_bfrag_swz(wt2 + 512);
    b2 = load_bfrag_swz(wt2 + 1024);
    b3 = load_bfrag_swz(wt2 + 1536);
    if (kt + 2 < IN_ / 32) {
      a0 = load_afrag(A, m0,      (kt + 2) * 32, IN_);
      a1 = load_afrag(A, m0 + 16, (kt + 2) * 32, IN_);
    }
    acc0 = wmma_f16(a0n, b0, acc0);
    acc1 = wmma_f16(a0n, b1, acc1);
    acc2 = wmma_f16(a0n, b2, acc2);
    acc3 = wmma_f16(a0n, b3, acc3);
    acc4 = wmma_f16(a1n, b0, acc4);
    acc5 = wmma_f16(a1n, b1, acc5);
    acc6 = wmma_f16(a1n, b2, acc6);
    acc7 = wmma_f16(a1n, b3, acc7);
  }
  int col = c_col();
#pragma unroll
  for (int r = 0; r < 8; ++r) {
    size_t row0 = (size_t)(m0 + c_row(r)) * R_;
    size_t row1 = (size_t)(m0 + 16 + c_row(r)) * R_;
    T[row0 +  0 + col] = (_Float16)acc0[r];
    T[row0 + 16 + col] = (_Float16)acc1[r];
    T[row0 + 32 + col] = (_Float16)acc2[r];
    T[row0 + 48 + col] = (_Float16)acc3[r];
    T[row1 +  0 + col] = (_Float16)acc4[r];
    T[row1 + 16 + col] = (_Float16)acc5[r];
    T[row1 + 32 + col] = (_Float16)acc6[r];
    T[row1 + 48 + col] = (_Float16)acc7[r];
  }
}

// ---------------- GEMM2: full projections ----------------
// MODE 0: Kmat = Tk@k_w2 + Tkp@kp_w2        (row-major f16 [M][1024])
// MODE 1: Vt   = (Tv@v_w2)^T per head       ([B*H*64][KVP] f16)
// MODE 2: Qh   = q + Tq@qp_w2               (row-major f16 [B*TQ][1024])
template <int MODE>
__global__ void __launch_bounds__(128) k_gemm2(const _Float16* __restrict__ T1,
                                               const _Float16* __restrict__ W2a,
                                               const _Float16* __restrict__ T2,
                                               const _Float16* __restrict__ W2b,
                                               const float* __restrict__ qin,
                                               _Float16* __restrict__ Out, int M) {
  int wid = (blockIdx.x * blockDim.x + threadIdx.x) >> 5;
  constexpr int NG = OUT_ / 64;     // 16 groups of 4 n-tiles
  int mtile = wid / NG, g = wid % NG;
  int m0 = mtile * 16;
  if (m0 >= M) return;
  v16h a0 = load_afrag(T1, m0, 0, R_);
  v16h a1 = load_afrag(T1, m0, 32, R_);
  v16h a2 = {}, a3 = {};
  if (MODE == 0) { a2 = load_afrag(T2, m0, 0, R_); a3 = load_afrag(T2, m0, 32, R_); }
  int nt0 = g * 4;
  v16h c0 = load_bfrag_swz(W2a + ((size_t)nt0 << 9));
  v16h c1 = load_bfrag_swz(W2a + ((size_t)(64 + nt0) << 9));
  v16h c2 = {}, c3 = {};
  if (MODE == 0) {
    c2 = load_bfrag_swz(W2b + ((size_t)nt0 << 9));
    c3 = load_bfrag_swz(W2b + ((size_t)(64 + nt0) << 9));
  }
#pragma unroll
  for (int j = 0; j < 4; ++j) {
    int nt = nt0 + j;
    v16h p0 = c0, p1 = c1, p2 = c2, p3 = c3;
    if (j < 3) {
      p0 = load_bfrag_swz(W2a + ((size_t)(nt + 1) << 9));
      p1 = load_bfrag_swz(W2a + ((size_t)(64 + nt + 1) << 9));
      if (MODE == 0) {
        p2 = load_bfrag_swz(W2b + ((size_t)(nt + 1) << 9));
        p3 = load_bfrag_swz(W2b + ((size_t)(64 + nt + 1) << 9));
      }
    }
    v8f acc = {};
    acc = wmma_f16(a0, c0, acc);
    acc = wmma_f16(a1, c1, acc);
    if (MODE == 0) {
      acc = wmma_f16(a2, c2, acc);
      acc = wmma_f16(a3, c3, acc);
    }
    int coln = nt * 16 + c_col();
    if (MODE == 1) {
      int b = m0 / KVP;
      int kvbase = (m0 % KVP) + ((laneid() >> 4) << 3);
      int hh = coln >> 6, dd = coln & 63;
      union { _Float16 h[8]; uint4 u; } pk;
#pragma unroll
      for (int r = 0; r < 8; ++r) pk.h[r] = (_Float16)acc[r];
      *reinterpret_cast<uint4*>(Out + (((size_t)(b * H_ + hh) * D_ + dd) * KVP + kvbase)) = pk.u;
    } else {
#pragma unroll
      for (int r = 0; r < 8; ++r) {
        int row = m0 + c_row(r);
        float val = acc[r];
        if (MODE == 2) {
          int b = row / TQ, t = row % TQ;
          int hh = coln >> 6, dd = coln & 63;
          val += qin[(((size_t)b * H_ + hh) * TQ + t) * D_ + dd];
        }
        Out[(size_t)row * OUT_ + coln] = (_Float16)val;
      }
    }
    c0 = p0; c1 = p1; c2 = p2; c3 = p3;
  }
}

// ---------------- flash attention ----------------
// 256 threads = 8 waves, 128 queries per block. K/V tiles staged into
// double-buffered LDS with GLOBAL_LOAD_ASYNC_TO_LDS_B128 (ASYNCcnt), drained
// with s_wait_asynccnt before the per-step barrier.
__global__ void __launch_bounds__(256) k_attn(const _Float16* __restrict__ Qh,
                                              const _Float16* __restrict__ Km,
                                              const _Float16* __restrict__ Vt,
                                              const float* __restrict__ mask,
                                              const float* __restrict__ resid,
                                              const float* __restrict__ gates,
                                              float* __restrict__ out) {
  __shared__ _Float16 sK[2][32 * 64];   // [kv 32][d 64]
  __shared__ _Float16 sV[2][64 * 32];   // [d 64][kv 32]
  __shared__ _Float16 sP[8][16 * 32];   // per-wave P tile
  int qb = blockIdx.x & 7;              // TQ/128 = 8 query blocks
  int bh = blockIdx.x >> 3;
  int h = bh % H_, b = bh / H_;
  int wave = threadIdx.x >> 5;
  int lane = laneid();
  int tid = threadIdx.x;
  int m0 = qb * 128 + wave * 16;        // query tile base within TQ
  const _Float16* Qb = Qh + (size_t)b * TQ * OUT_;
  const _Float16* Kb = Km + (size_t)b * KVP * OUT_;
  const _Float16* Vb = Vt + ((size_t)(b * H_ + h) * D_) * KVP;

  // staging thread mapping: K 32x64 halves (8 thr/row), V 64x32 (4 thr/row)
  int krow = tid >> 3, kcol = (tid & 7) * 8;
  int vrow = tid >> 2, vcol = (tid & 3) * 8;
  const _Float16* gK = Kb + (size_t)krow * OUT_ + h * D_ + kcol;
  const _Float16* gV = Vb + (size_t)vrow * KVP + vcol;
  unsigned lK[2] = { lds_off32(sK[0] + krow * 64 + kcol),
                     lds_off32(sK[1] + krow * 64 + kcol) };
  unsigned lV[2] = { lds_off32(sV[0] + vrow * 32 + vcol),
                     lds_off32(sV[1] + vrow * 32 + vcol) };

  v16h aq0 = load_afrag(Qb, m0, h * D_, OUT_);
  v16h aq1 = load_afrag(Qb, m0, h * D_ + 32, OUT_);
  v8f o0 = {}, o1 = {}, o2 = {}, o3 = {};
  float rm[8], rs[8];
#pragma unroll
  for (int r = 0; r < 8; ++r) { rm[r] = -1e30f; rs[r] = 0.f; }
  _Float16* myp = sP[wave];
  const float scale = 0.125f;           // 1/sqrt(64)

  // prime buffer 0 (async DMA, single hop)
  async_load_b128(lK[0], gK);
  async_load_b128(lV[0], gV);
  wait_asynccnt0();
  __syncthreads();

  constexpr int NSTEP = KVP / 32;       // 33
  for (int it = 0; it < NSTEP; ++it) {
    int kv0 = it * 32;
    int cur = it & 1;
    bool pf = (it + 1 < NSTEP);
    // issue next tile's async DMA into the alternate buffer
    if (pf) {
      async_load_b128(lK[cur ^ 1], gK + (size_t)(kv0 + 32) * OUT_);
      async_load_b128(lV[cur ^ 1], gV + kv0 + 32);
    }
    // mask values (global, overlap with WMMAs)
    float mv0[8], mv1[8];
    int col0 = kv0 + (lane & 15);
    int col1 = kv0 + 16 + (lane & 15);
#pragma unroll
    for (int r = 0; r < 8; ++r) {
      int t = m0 + c_row(r);
      mv0[r] = (col0 < KV_) ? mask[(size_t)t * KV_ + col0] : -1e30f;
      mv1[r] = (col1 < KV_) ? mask[(size_t)t * KV_ + col1] : -1e30f;
    }
    const _Float16* curK = sK[cur];
    const _Float16* curV = sV[cur];
    // scores: 16 queries x 32 keys (two 16x16 C tiles), K frags from LDS
    v8f s0 = {}, s1 = {};
    s0 = wmma_f16(aq0, load_bfrag_t(curK, 0,  0,  64), s0);
    s0 = wmma_f16(aq1, load_bfrag_t(curK, 0,  32, 64), s0);
    s1 = wmma_f16(aq0, load_bfrag_t(curK, 16, 0,  64), s1);
    s1 = wmma_f16(aq1, load_bfrag_t(curK, 16, 32, 64), s1);
    // online softmax per accumulator row (reduce within each 16-lane half)
#pragma unroll
    for (int r = 0; r < 8; ++r) {
      s0[r] = s0[r] * scale + mv0[r];
      s1[r] = s1[r] * scale + mv1[r];
      float mx = fmaxf(s0[r], s1[r]);
#pragma unroll
      for (int d = 1; d < 16; d <<= 1) mx = fmaxf(mx, __shfl_xor(mx, d, 16));
      float nm = fmaxf(rm[r], mx);
      float corr = __expf(rm[r] - nm);
      rm[r] = nm;
      float p0 = __expf(s0[r] - nm);
      float p1 = __expf(s1[r] - nm);
      float ps = p0 + p1;
#pragma unroll
      for (int d = 1; d < 16; d <<= 1) ps += __shfl_xor(ps, d, 16);
      rs[r] = rs[r] * corr + ps;
      o0[r] *= corr; o1[r] *= corr; o2[r] *= corr; o3[r] *= corr;
      int rr = c_row(r);
      myp[rr * 32 + (lane & 15)]      = (_Float16)p0;   // DS in-order per wave
      myp[rr * 32 + 16 + (lane & 15)] = (_Float16)p1;
    }
    // O += P (16x32) @ V (32x64); V^T frags from LDS
    v16h ap = load_afrag(myp, 0, 0, 32);
    o0 = wmma_f16(ap, load_bfrag_t(curV, 0,  0, 32), o0);
    o1 = wmma_f16(ap, load_bfrag_t(curV, 16, 0, 32), o1);
    o2 = wmma_f16(ap, load_bfrag_t(curV, 32, 0, 32), o2);
    o3 = wmma_f16(ap, load_bfrag_t(curV, 48, 0, 32), o3);
    // drain this wave's async DMA before the barrier so the next step may read
    if (pf) wait_asynccnt0();
    __syncthreads();
  }
  // epilogue: normalize, gate, residual
  float g = gates[0];
  int coln = lane & 15;
#pragma unroll
  for (int r = 0; r < 8; ++r) {
    int t = m0 + c_row(r);
    float inv = 1.f / rs[r];
    size_t base = ((size_t)b * TQ + t) * OUT_ + h * D_ + coln;
    out[base]      = o0[r] * inv * g + resid[base];
    out[base + 16] = o1[r] * inv * g + resid[base + 16];
    out[base + 32] = o2[r] * inv * g + resid[base + 32];
    out[base + 48] = o3[r] * inv * g + resid[base + 48];
  }
}

// ---------------- host launcher ----------------
extern "C" void kernel_launch(void* const* d_in, const int* in_sizes, int n_in,
                              void* d_out, int out_size, void* d_ws, size_t ws_size,
                              hipStream_t stream) {
  (void)in_sizes; (void)n_in; (void)out_size; (void)ws_size;
  const float* pe        = (const float*)d_in[0];
  const float* attn_out  = (const float*)d_in[1];
  const float* q         = (const float*)d_in[2];
  const float* kv_query  = (const float*)d_in[3];
  const float* attn_mask = (const float*)d_in[4];
  const float* prompt    = (const float*)d_in[5];
  const float* gates     = (const float*)d_in[6];
  const float* k_w1  = (const float*)d_in[7];
  const float* k_w2  = (const float*)d_in[8];
  const float* v_w1  = (const float*)d_in[9];
  const float* v_w2  = (const float*)d_in[10];
  const float* kp_w1 = (const float*)d_in[11];
  const float* kp_w2 = (const float*)d_in[12];
  const float* qp_w1 = (const float*)d_in[13];
  const float* qp_w2 = (const float*)d_in[14];
  const int* indices_a = (const int*)d_in[15];
  const int* indices_b = (const int*)d_in[16];
  const int* task_idx  = (const int*)d_in[17];
  float* out = (float*)d_out;

  // workspace carve-up (halves); total ~54.6 MB, L2-resident (192 MB)
  _Float16* ws = (_Float16*)d_ws;
  size_t off = 0;
  auto alloc = [&](size_t n) { _Float16* p = ws + off; off += n; return p; };
  _Float16* Xkv  = alloc((size_t)B_ * KVP * IN_);
  _Float16* Xpos = alloc((size_t)B_ * KVP * OUT_);
  _Float16* Xpea = alloc((size_t)B_ * TQ * OUT_);
  _Float16* Wk1s  = alloc((size_t)IN_ * R_);
  _Float16* Wv1s  = alloc((size_t)IN_ * R_);
  _Float16* Wkp1s = alloc((size_t)IN_ * R_);
  _Float16* Wqp1s = alloc((size_t)IN_ * R_);
  _Float16* Wk2s  = alloc((size_t)R_ * OUT_);
  _Float16* Wv2s  = alloc((size_t)R_ * OUT_);
  _Float16* Wkp2s = alloc((size_t)R_ * OUT_);
  _Float16* Wqp2s = alloc((size_t)R_ * OUT_);
  _Float16* Tk  = alloc((size_t)B_ * KVP * R_);
  _Float16* Tv  = alloc((size_t)B_ * KVP * R_);
  _Float16* Tkp = alloc((size_t)B_ * KVP * R_);
  _Float16* Tq  = alloc((size_t)B_ * TQ * R_);
  _Float16* Kmat = alloc((size_t)B_ * KVP * OUT_);
  _Float16* Vt   = alloc((size_t)B_ * H_ * D_ * KVP);
  _Float16* Qh   = alloc((size_t)B_ * TQ * OUT_);

  const int WE = (IN_ * R_ + 255) / 256;   // 65536 elems per weight
  k_swizzle_w<<<WE, 256, 0, stream>>>(k_w1,  Wk1s,  IN_, R_);
  k_swizzle_w<<<WE, 256, 0, stream>>>(v_w1,  Wv1s,  IN_, R_);
  k_swizzle_w<<<WE, 256, 0, stream>>>(kp_w1, Wkp1s, IN_, R_);
  k_swizzle_w<<<WE, 256, 0, stream>>>(qp_w1, Wqp1s, IN_, R_);
  k_swizzle_w<<<WE, 256, 0, stream>>>(k_w2,  Wk2s,  R_, OUT_);
  k_swizzle_w<<<WE, 256, 0, stream>>>(v_w2,  Wv2s,  R_, OUT_);
  k_swizzle_w<<<WE, 256, 0, stream>>>(kp_w2, Wkp2s, R_, OUT_);
  k_swizzle_w<<<WE, 256, 0, stream>>>(qp_w2, Wqp2s, R_, OUT_);

  k_gather_xkv <<<B_ * KVP, 256, 0, stream>>>(prompt, kv_query, task_idx, Xkv);
  k_gather_xpos<<<B_ * KVP, 256, 0, stream>>>(pe, indices_b, Xpos);
  k_gather_xpea<<<B_ * TQ,  256, 0, stream>>>(pe, indices_a, Xpea);

  // GEMM1: 32 rows per wave -> waves = M/32, 4 waves per block
  k_gemm1<<<(B_ * KVP) / 128, 128, 0, stream>>>(Xkv,  Wk1s,  Tk, B_ * KVP);
  k_gemm1<<<(B_ * KVP) / 128, 128, 0, stream>>>(Xkv,  Wv1s,  Tv, B_ * KVP);
  k_gemm1<<<(B_ * KVP) / 128, 128, 0, stream>>>(Xpos, Wkp1s, Tkp, B_ * KVP);
  k_gemm1<<<(B_ * TQ)  / 128, 128, 0, stream>>>(Xpea, Wqp1s, Tq, B_ * TQ);

  int wavesK = (B_ * KVP / 16) * (OUT_ / 64);  // 4224
  int wavesQ = (B_ * TQ  / 16) * (OUT_ / 64);  // 4096
  k_gemm2<0><<<wavesK / 4, 128, 0, stream>>>(Tk, Wk2s, Tkp, Wkp2s, nullptr, Kmat, B_ * KVP);
  k_gemm2<1><<<wavesK / 4, 128, 0, stream>>>(Tv, Wv2s, nullptr, nullptr, nullptr, Vt, B_ * KVP);
  k_gemm2<2><<<wavesQ / 4, 128, 0, stream>>>(Tq, Wqp2s, nullptr, nullptr, q, Qh, B_ * TQ);

  // attention: 128 queries per 256-thread block
  k_attn<<<B_ * H_ * (TQ / 128), 256, 0, stream>>>(Qh, Kmat, Vt, attn_mask,
                                                   attn_out, gates, out);
}
